// icl_loss_22084721836062
// MI455X (gfx1250) — matrix operand order repaired and checked by reading, not compiled, not run on hardware
//
#include <hip/hip_runtime.h>
#include <hip/hip_bf16.h>
#include <math.h>

// ---------------------------------------------------------------------------
// Problem constants (from reference)
// ---------------------------------------------------------------------------
#define DIM      768
#define BATCH    4096
#define INV_TAU  20.0f      // 1 / 0.05
// ALPHA = 0.5 is folded into the finalize kernel.

typedef _Float16 v16h __attribute__((ext_vector_type(16)));
typedef _Float16 v8h  __attribute__((ext_vector_type(8)));
typedef float    v8f  __attribute__((ext_vector_type(8)));

// ---------------------------------------------------------------------------
// Kernel 0: zero the row/col exp-sum accumulators
// ---------------------------------------------------------------------------
__global__ void zero_kernel(float* p, int n) {
    int i = blockIdx.x * blockDim.x + threadIdx.x;
    if (i < n) p[i] = 0.0f;
}

// ---------------------------------------------------------------------------
// Kernel 1: gather + L2-normalize (f32 math) -> f16 rows; diag logit in f32.
// One block per batch element; 256 threads; 768 = 3*256 elements per row.
// ---------------------------------------------------------------------------
__global__ __launch_bounds__(256)
void normalize_kernel(const float* __restrict__ emb,
                      const int*   __restrict__ links,
                      _Float16* __restrict__ h1,
                      _Float16* __restrict__ h2,
                      float* __restrict__ diag) {
    const int i = blockIdx.x;
    const int t = threadIdx.x;
    const int i0 = links[2 * i + 0];
    const int i1 = links[2 * i + 1];
    const float* r0 = emb + (size_t)i0 * DIM;
    const float* r1 = emb + (size_t)i1 * DIM;

    float x0[3], x1[3];
    float s0 = 0.0f, s1 = 0.0f;
#pragma unroll
    for (int j = 0; j < 3; ++j) {
        x0[j] = r0[t + 256 * j];
        x1[j] = r1[t + 256 * j];
        s0 += x0[j] * x0[j];
        s1 += x1[j] * x1[j];
    }

    __shared__ float red0[256];
    __shared__ float red1[256];
    red0[t] = s0; red1[t] = s1;
    __syncthreads();
    for (int off = 128; off > 0; off >>= 1) {
        if (t < off) { red0[t] += red0[t + off]; red1[t] += red1[t + off]; }
        __syncthreads();
    }
    const float inv0 = rsqrtf(red0[0]);
    const float inv1 = rsqrtf(red1[0]);
    __syncthreads();   // before reusing red0

    float d = 0.0f;
#pragma unroll
    for (int j = 0; j < 3; ++j) {
        const float a = x0[j] * inv0;
        const float b = x1[j] * inv1;
        h1[(size_t)i * DIM + t + 256 * j] = (_Float16)a;
        h2[(size_t)i * DIM + t + 256 * j] = (_Float16)b;
        d += a * b;
    }
    red0[t] = d;
    __syncthreads();
    for (int off = 128; off > 0; off >>= 1) {
        if (t < off) red0[t] += red0[t + off];
        __syncthreads();
    }
    if (t == 0) diag[i] = red0[0] * INV_TAU;
}

// ---------------------------------------------------------------------------
// Kernel 2: fused GEMM + exp + row/col reduction.
// Each wave computes a 64x64 tile of S = A * B^T via v_wmma_f32_16x16x32_f16
// with 4x4 register blocking: 16 WMMAs per K-step fed by 16 b128 loads
// (1 load per WMMA, 32 FLOP per L2 byte -> compute-bound, not L2-bound).
// Exponentiation happens in registers (no max pass needed: |logit| <= 20),
// diagonal masked to exact 0, then per-row sums (and per-column sums for the
// cross matrix, since logits_ba = logits_ab^T) are reduced with lane shuffles
// and accumulated with one global_atomic_add_f32 per row/column per wave.
//
// WMMA wave32 f16 fragment layouts as used here (A/B use contiguous K, not the ISA table order):
//   A 16x32: lane l<16 -> row l, elems 0..15 = K 0..15 (contiguous)
//            lane l>=16 -> row l-16, K 16..31
//   B 32x16: lane l<16 -> col l, elems 0..15 = K 0..15 (contiguous)
//            lane l>=16 -> col l-16, K 16..31
//   C/D 16x16: VGPR v, lane l -> M = v + (l>=16 ? 8 : 0), N = l & 15
// ---------------------------------------------------------------------------
__global__ __launch_bounds__(256)
void gemm_exp_kernel(const _Float16* __restrict__ Amat,
                     const _Float16* __restrict__ Bmat,
                     float* __restrict__ rowAcc,
                     float* __restrict__ colAcc,
                     int maskDiag) {
    const int lane  = threadIdx.x & 31;
    const int wave  = threadIdx.x >> 5;
    const int gw    = blockIdx.x * 8 + wave;      // global wave-tile id
    const int tileM = gw >> 6;                    // 4096/64 = 64 tiles per dim
    const int tileN = gw & 63;
    const int m0 = tileM * 64;
    const int n0 = tileN * 64;
    const int lhalf = lane >> 4;                  // 0 or 1
    const int l15   = lane & 15;

    const _Float16* aRow[4];
    const _Float16* bRow[4];
#pragma unroll
    for (int mi = 0; mi < 4; ++mi)
        aRow[mi] = Amat + (size_t)(m0 + mi * 16 + l15) * DIM;
#pragma unroll
    for (int ni = 0; ni < 4; ++ni)
        bRow[ni] = Bmat + (size_t)(n0 + ni * 16 + l15) * DIM;

    const int aoff = lhalf * 16;   // A half-wave K shift (elements)
    const int boff = lhalf * 16;   // B half-wave K shift (elements)

    v8f c[4][4];
#pragma unroll
    for (int mi = 0; mi < 4; ++mi)
#pragma unroll
        for (int ni = 0; ni < 4; ++ni) c[mi][ni] = (v8f){};

    for (int k0 = 0; k0 < DIM; k0 += 32) {
        v16h a[4], b[4];
#pragma unroll
        for (int mi = 0; mi < 4; ++mi) {
            v8h lo = *(const v8h*)(aRow[mi] + k0 + aoff);
            v8h hi = *(const v8h*)(aRow[mi] + k0 + aoff + 8);
            a[mi] = __builtin_shufflevector(lo, hi, 0,1,2,3,4,5,6,7,8,9,10,11,12,13,14,15);
        }
#pragma unroll
        for (int ni = 0; ni < 4; ++ni)
            b[ni] = *(const v16h*)(bRow[ni] + k0 + boff);   // 32B contiguous

#pragma unroll
        for (int mi = 0; mi < 4; ++mi)
#pragma unroll
            for (int ni = 0; ni < 4; ++ni)
                c[mi][ni] = __builtin_amdgcn_wmma_f32_16x16x32_f16(
                    false, a[mi], false, b[ni], (short)0, c[mi][ni], false, false);
    }

    // Exponentiate IN PLACE (diag masked -> exact 0, matching exp(x - 1e9))
#pragma unroll
    for (int mi = 0; mi < 4; ++mi)
#pragma unroll
        for (int ni = 0; ni < 4; ++ni)
#pragma unroll
            for (int v = 0; v < 8; ++v) {
                const int m = m0 + mi * 16 + v + lhalf * 8;
                const int n = n0 + ni * 16 + l15;
                float ee = __expf(c[mi][ni][v] * INV_TAU);
                if (maskDiag && (m == n)) ee = 0.0f;
                c[mi][ni][v] = ee;
            }

    // Row sums: for row M=v(+8), the 16 N-values live across one 16-lane half.
#pragma unroll
    for (int mi = 0; mi < 4; ++mi)
#pragma unroll
        for (int v = 0; v < 8; ++v) {
            float rv = c[mi][0][v] + c[mi][1][v] + c[mi][2][v] + c[mi][3][v];
            rv += __shfl_xor(rv, 1);
            rv += __shfl_xor(rv, 2);
            rv += __shfl_xor(rv, 4);
            rv += __shfl_xor(rv, 8);
            if (l15 == 0)
                atomicAdd(&rowAcc[m0 + mi * 16 + v + lhalf * 8], rv);
        }

    // Column sums (cross matrix only): each lane owns column N=l15 for 32 rows
    // of its half; the other 32 rows sit in the partner lane (xor 16).
    if (colAcc != nullptr) {
#pragma unroll
        for (int ni = 0; ni < 4; ++ni) {
            float cv = 0.0f;
#pragma unroll
            for (int mi = 0; mi < 4; ++mi)
#pragma unroll
                for (int v = 0; v < 8; ++v) cv += c[mi][ni][v];
            cv += __shfl_xor(cv, 16);
            if (lhalf == 0)
                atomicAdd(&colAcc[n0 + ni * 16 + l15], cv);
        }
    }
}

// ---------------------------------------------------------------------------
// Kernel 3: finalize.  loss = mean_i( 0.5*log(sum_a) + 0.5*log(sum_b) - diag )
// ---------------------------------------------------------------------------
__global__ __launch_bounds__(256)
void final_kernel(const float* __restrict__ sum_a,
                  const float* __restrict__ sum_b,
                  const float* __restrict__ diag,
                  float* __restrict__ out) {
    __shared__ float red[256];
    const int t = threadIdx.x;
    float acc = 0.0f;
    for (int i = t; i < BATCH; i += 256)
        acc += 0.5f * (logf(sum_a[i]) + logf(sum_b[i])) - diag[i];
    red[t] = acc;
    __syncthreads();
    for (int off = 128; off > 0; off >>= 1) {
        if (t < off) red[t] += red[t + off];
        __syncthreads();
    }
    if (t == 0) out[0] = red[0] * (1.0f / (float)BATCH);
}

// ---------------------------------------------------------------------------
// Host-side launcher
// ---------------------------------------------------------------------------
extern "C" void kernel_launch(void* const* d_in, const int* in_sizes, int n_in,
                              void* d_out, int out_size, void* d_ws, size_t ws_size,
                              hipStream_t stream) {
    (void)in_sizes; (void)n_in; (void)out_size; (void)ws_size;

    const float* emb   = (const float*)d_in[0];   // (100000, 768) f32
    const int*   links = (const int*)d_in[1];     // (4096, 2)   int32
    float*       out   = (float*)d_out;           // scalar f32

    // Workspace carve-out (~12.7 MB)
    char* ws = (char*)d_ws;
    const size_t hbytes = (size_t)BATCH * DIM * sizeof(_Float16);   // 6.29 MB
    _Float16* h1    = (_Float16*)(ws);
    _Float16* h2    = (_Float16*)(ws + hbytes);
    float*    sum_a = (float*)(ws + 2 * hbytes);
    float*    sum_b = sum_a + BATCH;
    float*    diag  = sum_b + BATCH;

    // 0) zero exp-sum accumulators (sum_a and sum_b are contiguous)
    zero_kernel<<<(2 * BATCH + 255) / 256, 256, 0, stream>>>(sum_a, 2 * BATCH);

    // 1) gather + normalize + diag logits
    normalize_kernel<<<BATCH, 256, 0, stream>>>(emb, links, h1, h2, diag);

    // 2) fused GEMM + exp + reductions
    //    4096/64 = 64 wave-tiles per dim -> 4096 wave-tiles -> 512 blocks of 8 waves
    const int gemm_blocks = (64 * 64) / 8;
    gemm_exp_kernel<<<gemm_blocks, 256, 0, stream>>>(h1, h1, sum_a, nullptr, 1); // logits_aa (masked)
    gemm_exp_kernel<<<gemm_blocks, 256, 0, stream>>>(h2, h2, sum_b, nullptr, 1); // logits_bb (masked)
    gemm_exp_kernel<<<gemm_blocks, 256, 0, stream>>>(h1, h2, sum_a, sum_b, 0);   // logits_ab (+ ba via cols)

    // 3) finalize scalar loss
    final_kernel<<<1, 256, 0, stream>>>(sum_a, sum_b, diag, out);
}
